// Mamba2Layer_18184891532028
// MI455X (gfx1250) — compile-verified, matrix operand order checked by reading, not run on hardware
//
#include <hip/hip_runtime.h>
#include <hip/hip_fp16.h>
#include <math.h>

#define D_MODEL   1024
#define D_STATE   64
#define D_CONV    4
#define D_INNER   2048
#define DT_RANK   64
#define B_SZ      2
#define SEQLEN    2048
#define NTOK      (B_SZ * SEQLEN)            // 4096 tokens
#define XPROJ_OUT (DT_RANK + 2 * D_STATE)    // 192

typedef __attribute__((ext_vector_type(16))) _Float16 v16h;
typedef __attribute__((ext_vector_type(8)))  float    v8f;
typedef __attribute__((ext_vector_type(4)))  float    v4f;
typedef __attribute__((ext_vector_type(4)))  int      v4i;

#define AS1 __attribute__((address_space(1)))
#define AS3 __attribute__((address_space(3)))

// --------------------------- async copy plumbing ---------------------------
#if defined(__has_builtin)
#  if __has_builtin(__builtin_amdgcn_global_load_async_to_lds_b128)
#    define HAS_ASYNC_LDS 1
#  endif
#endif
#ifndef HAS_ASYNC_LDS
#  define HAS_ASYNC_LDS 0
#endif

#if HAS_ASYNC_LDS
#  if __has_builtin(__builtin_amdgcn_s_wait_asynccnt)
#    define WAIT_ASYNC(n) __builtin_amdgcn_s_wait_asynccnt(n)
#  else
#    define WAIT_ASYNC(n) asm volatile("s_wait_asynccnt %0" :: "n"(n) : "memory")
#  endif
#else
#  define WAIT_ASYNC(n) ((void)0)
#endif

// 16-byte global -> LDS copy (async on gfx1250, sync fallback otherwise)
__device__ __forceinline__ void cp16_g2l(const void* g, void* l) {
#if HAS_ASYNC_LDS
  __builtin_amdgcn_global_load_async_to_lds_b128(
      (AS1 v4i*)const_cast<void*>(g), (AS3 v4i*)l, 0, 0);
#else
  *(v4f*)l = *(const v4f*)g;
#endif
}

// ---------------------------------------------------------------------------
// WMMA fragment helpers (CDNA5 16-bit A-matrix 16x32 layout, ISA 7.12.2).
// Same gather for A (lane->M) and B (lane->N): both operands of C = X * W^T
// index K contiguously along rows.
// LDS tile rows: 32 halves of K payload padded to 40 halves (80 B) to break
// the 4-way bank conflict of 64 B-strided b128 reads.
// ---------------------------------------------------------------------------
#define LROW 40

__device__ __forceinline__ v16h lds_frag(const _Float16* base) {
  int lane = threadIdx.x & 31;
  const _Float16* q = base + (lane & 15) * LROW + ((lane & 16) ? 8 : 0);
  union { v16h h; v4f f[2]; } u;
  u.f[0] = *(const v4f*)q;          // K 0..7   (or 8..15)
  u.f[1] = *(const v4f*)(q + 16);   // K 16..23 (or 24..31)
  return u.h;
}

// 32-bit C/D 16x16 layout: lane<16 -> N=lane, M=row; lane>=16 -> M=row+8.
__device__ __forceinline__ void store_tile(float* __restrict__ c, int ldc, v8f acc) {
  int lane = threadIdx.x & 31;
  float* q = c + (size_t)((lane & 16) ? 8 : 0) * ldc + (lane & 15);
#pragma unroll
  for (int r = 0; r < 8; ++r) q[(size_t)r * ldc] = acc[r];
}

// Stage ROWS rows x 64 B (one K=32 slab) from global (row stride ld halves)
// into LDS (row stride LROW halves). 4 chunks of 16 B per row.
// Branchless: blockDim is exactly 256, ROWS*4 is a multiple of 256, so each
// thread does a fixed number of copies (no EXEC-masked loop tail).
template<int ROWS>
__device__ __forceinline__ void stage_tile(const _Float16* __restrict__ g0, int ld,
                                           _Float16* l0) {
  constexpr int ITER = (ROWS * 4) / 256;     // 2 for 128 rows, 1 for 64 rows
#pragma unroll
  for (int it = 0; it < ITER; ++it) {
    int c = threadIdx.x + it * 256;
    int r = c >> 2, s = c & 3;
    cp16_g2l(g0 + (size_t)r * ld + s * 8, l0 + r * LROW + s * 8);
  }
}

// ---------------------------------------------------------------------------
// C[M,N] = A[M,K] * B[N,K]^T, fp16 in / fp32 accumulate.
// 256 threads = 8 waves (4 M x 2 N). Block tile 128 x BN, wave tile 32 x BN/2.
// Double-buffered async LDS staging of both operands.
// ---------------------------------------------------------------------------
template<int BN>   // 128 (wide) or 64 (for the 192-col x_proj GEMM)
__global__ __launch_bounds__(256) void wmma_gemm_xwt(
    const _Float16* __restrict__ A, const _Float16* __restrict__ B,
    float* __restrict__ C, int K, int lda, int ldb, int ldc) {
  constexpr int WN  = BN / 32;        // 16-wide subtiles per wave in N
  constexpr int NST = 2 + BN / 64;    // async instrs per wave per stage
  __shared__ _Float16 sA[2][128 * LROW];
  __shared__ _Float16 sB[2][BN * LROW];

  int wave = threadIdx.x >> 5;
  int wm = wave & 3, wn = wave >> 2;
  const _Float16* Ag = A + (size_t)(blockIdx.x * 128) * lda;
  const _Float16* Bg = B + (size_t)blockIdx.y * BN * ldb;

  v8f zero = {};
  v8f acc[2][WN];
#pragma unroll
  for (int i = 0; i < 2; ++i)
#pragma unroll
    for (int j = 0; j < WN; ++j) acc[i][j] = zero;

  stage_tile<128>(Ag, lda, sA[0]);
  stage_tile<BN>(Bg, ldb, sB[0]);
  int buf = 0;
  for (int k = 0; k < K; k += 32, buf ^= 1) {
    if (k + 32 < K) {                       // prefetch next slab into other buffer
      stage_tile<128>(Ag + k + 32, lda, sA[buf ^ 1]);
      stage_tile<BN>(Bg + k + 32, ldb, sB[buf ^ 1]);
      WAIT_ASYNC(NST);                      // retire previous stage only
    } else {
      WAIT_ASYNC(0);
    }
    __syncthreads();                        // staged data visible to all waves
    v16h a0 = lds_frag(&sA[buf][(wm * 32) * LROW]);
    v16h a1 = lds_frag(&sA[buf][(wm * 32 + 16) * LROW]);
#pragma unroll
    for (int j = 0; j < WN; ++j) {
      v16h b = lds_frag(&sB[buf][(wn * (BN / 2) + j * 16) * LROW]);
      acc[0][j] = __builtin_amdgcn_wmma_f32_16x16x32_f16(false, a0, false, b, (short)0, acc[0][j], false, false);
      acc[1][j] = __builtin_amdgcn_wmma_f32_16x16x32_f16(false, a1, false, b, (short)0, acc[1][j], false, false);
    }
    __syncthreads();                        // all reads done -> buffer reusable
  }

  int m0 = blockIdx.x * 128 + wm * 32;
  int n0 = blockIdx.y * BN + wn * (BN / 2);
#pragma unroll
  for (int j = 0; j < WN; ++j) {
    store_tile(C + (size_t)m0 * ldc + n0 + j * 16,        ldc, acc[0][j]);
    store_tile(C + (size_t)(m0 + 16) * ldc + n0 + j * 16, ldc, acc[1][j]);
  }
}

// ---------------------------------------------------------------------------
__global__ void cvt_f32_f16_kernel(const float* __restrict__ s,
                                   _Float16* __restrict__ d, int n) {
  int i = blockIdx.x * blockDim.x + threadIdx.x;
  if (i < n) d[i] = (_Float16)s[i];
}

// Causal depthwise conv1d (k=4) + bias + SiLU over x = xz[..., :D_INNER].
__global__ __launch_bounds__(256) void conv_silu_kernel(
    const float* __restrict__ xz, const float* __restrict__ w,
    const float* __restrict__ bias, float* __restrict__ xo,
    _Float16* __restrict__ xo16) {
  int d = blockIdx.x * 256 + threadIdx.x;
  int l = blockIdx.y, b = blockIdx.z;
  size_t base = (size_t)b * SEQLEN * (2 * D_INNER);
  float acc = bias[d];
#pragma unroll
  for (int j = 0; j < D_CONV; ++j) {
    int li = l - (D_CONV - 1) + j;
    if (li >= 0) acc += w[d * D_CONV + j] * xz[base + (size_t)li * (2 * D_INNER) + d];
  }
  float s = acc / (1.f + __expf(-acc));
  size_t o = ((size_t)(b * SEQLEN + l)) * D_INNER + d;
  xo[o] = s;
  xo16[o] = (_Float16)s;
}

__global__ void softplus_bias_kernel(float* __restrict__ dt,
                                     const float* __restrict__ bias, int n) {
  for (int i = blockIdx.x * blockDim.x + threadIdx.x; i < n;
       i += gridDim.x * blockDim.x) {
    float v = dt[i] + bias[i & (D_INNER - 1)];
    dt[i] = (v > 20.f) ? v : log1pf(__expf(v));
  }
}

// ---------------------------------------------------------------------------
// Selective scan: one wave32 per (b,d) channel, 2 states/lane (n, n+32).
// B_t/C_t 32-step chunks double-buffered in LDS via async copies; chunk t+1
// streams in while the wave iterates chunk t (hides latency on the serial path).
// ---------------------------------------------------------------------------
__global__ __launch_bounds__(256) void scan_kernel(
    const float* __restrict__ xdbl, const float* __restrict__ dt,
    const float* __restrict__ u, const float* __restrict__ A_log,
    float* __restrict__ y) {
  __shared__ float sB[2][32 * D_STATE];
  __shared__ float sC[2][32 * D_STATE];
  int tid = threadIdx.x;
  int wave = tid >> 5, lane = tid & 31;
  int d = blockIdx.x * 8 + wave;
  int b = blockIdx.y;
  int n0 = lane, n1 = lane + 32;
  float a0 = -__expf(A_log[(size_t)d * D_STATE + n0]);
  float a1 = -__expf(A_log[(size_t)d * D_STATE + n1]);
  float h0 = 0.f, h1 = 0.f;

  auto stage = [&](int bufi, int t0) {       // 4 async instrs per wave
#pragma unroll
    for (int it = 0; it < 2; ++it) {
      int c = tid + it * 256;                // 512 x 16B chunks each for B and C
      int r = c >> 4, s = c & 15;
      const float* gB = xdbl + (size_t)(b * SEQLEN + t0 + r) * XPROJ_OUT + DT_RANK + s * 4;
      cp16_g2l(gB,           &sB[bufi][r * D_STATE + s * 4]);
      cp16_g2l(gB + D_STATE, &sC[bufi][r * D_STATE + s * 4]);
    }
  };

  stage(0, 0);
  for (int t0 = 0; t0 < SEQLEN; t0 += 32) {
    int buf = (t0 >> 5) & 1;
    if (t0 + 32 < SEQLEN) { stage(buf ^ 1, t0 + 32); WAIT_ASYNC(4); }
    else                  { WAIT_ASYNC(0); }
    size_t rb = ((size_t)(b * SEQLEN + t0 + lane)) * D_INNER + d;
    float dtq = dt[rb];   // lane t' holds step t0+t' scalars, shuffled below
    float uq  = u[rb];
    __syncthreads();
#pragma unroll 4
    for (int tt = 0; tt < 32; ++tt) {
      float dt_t = __shfl(dtq, tt, 32);
      float u_t  = __shfl(uq,  tt, 32);
      float bb0 = sB[buf][tt * D_STATE + n0], bb1 = sB[buf][tt * D_STATE + n1];
      float cc0 = sC[buf][tt * D_STATE + n0], cc1 = sC[buf][tt * D_STATE + n1];
      float dtu = dt_t * u_t;
      h0 = __expf(dt_t * a0) * h0 + dtu * bb0;
      h1 = __expf(dt_t * a1) * h1 + dtu * bb1;
      float part = h0 * cc0 + h1 * cc1;
#pragma unroll
      for (int off = 16; off > 0; off >>= 1) part += __shfl_xor(part, off, 32);
      if (lane == 0)
        y[((size_t)(b * SEQLEN + t0 + tt)) * D_INNER + d] = part;
    }
    __syncthreads();      // all reads done -> buffer reusable
  }
}

// (y + x*D) * silu(z) -> f16 for the out_proj GEMM.
__global__ void gate_kernel(const float* __restrict__ y, const float* __restrict__ xc,
                            const float* __restrict__ Dv, const float* __restrict__ xz,
                            _Float16* __restrict__ g16, int n) {
  for (int i = blockIdx.x * blockDim.x + threadIdx.x; i < n;
       i += gridDim.x * blockDim.x) {
    int d = i & (D_INNER - 1);
    int row = i >> 11;                          // D_INNER == 2048
    float z = xz[(size_t)row * (2 * D_INNER) + D_INNER + d];
    float yy = y[i] + xc[i] * Dv[d];
    float sz = z / (1.f + __expf(-z));
    g16[i] = (_Float16)(yy * sz);
  }
}

// ---------------------------------------------------------------------------
extern "C" void kernel_launch(void* const* d_in, const int* in_sizes, int n_in,
                              void* d_out, int out_size, void* d_ws, size_t ws_size,
                              hipStream_t stream) {
  (void)in_sizes; (void)n_in; (void)out_size; (void)ws_size;
  const float* hs    = (const float*)d_in[0];
  const float* inw   = (const float*)d_in[1];
  const float* convw = (const float*)d_in[2];
  const float* convb = (const float*)d_in[3];
  const float* xpw   = (const float*)d_in[4];
  const float* dtw   = (const float*)d_in[5];
  const float* dtb   = (const float*)d_in[6];
  const float* Alog  = (const float*)d_in[7];
  const float* Dv    = (const float*)d_in[8];
  const float* outw  = (const float*)d_in[9];
  float* out = (float*)d_out;

  char* ws = (char*)d_ws;
  size_t off = 0;
  auto alloc = [&](size_t bytes) -> void* {
    void* p = ws + off;
    off = (off + bytes + 255) & ~(size_t)255;
    return p;
  };
  _Float16* hs16   = (_Float16*)alloc((size_t)NTOK * D_MODEL * 2);
  _Float16* inw16  = (_Float16*)alloc((size_t)(2 * D_INNER) * D_MODEL * 2);
  _Float16* xpw16  = (_Float16*)alloc((size_t)XPROJ_OUT * D_INNER * 2);
  _Float16* dtw16  = (_Float16*)alloc((size_t)D_INNER * DT_RANK * 2);
  _Float16* outw16 = (_Float16*)alloc((size_t)D_MODEL * D_INNER * 2);
  float*    xz     = (float*)   alloc((size_t)NTOK * 2 * D_INNER * 4);
  float*    xc     = (float*)   alloc((size_t)NTOK * D_INNER * 4);
  _Float16* xc16   = (_Float16*)alloc((size_t)NTOK * D_INNER * 2);
  float*    xdbl   = (float*)   alloc((size_t)NTOK * XPROJ_OUT * 4);
  _Float16* xdbl16 = (_Float16*)alloc((size_t)NTOK * XPROJ_OUT * 2);
  float*    dtraw  = (float*)   alloc((size_t)NTOK * D_INNER * 4);
  float*    yb     = (float*)   alloc((size_t)NTOK * D_INNER * 4);
  _Float16* g16    = (_Float16*)alloc((size_t)NTOK * D_INNER * 2);

  auto cvt = [&](const float* s, _Float16* d, size_t n) {
    cvt_f32_f16_kernel<<<dim3((unsigned)((n + 255) / 256)), 256, 0, stream>>>(s, d, (int)n);
  };
  // 1) weight / activation down-converts
  cvt(hs,   hs16,   (size_t)NTOK * D_MODEL);
  cvt(inw,  inw16,  (size_t)(2 * D_INNER) * D_MODEL);
  cvt(xpw,  xpw16,  (size_t)XPROJ_OUT * D_INNER);
  cvt(dtw,  dtw16,  (size_t)D_INNER * DT_RANK);
  cvt(outw, outw16, (size_t)D_MODEL * D_INNER);

  // 2) in_proj: xz[4096,4096] = hs[4096,1024] @ in_proj_w^T
  wmma_gemm_xwt<128><<<dim3(NTOK / 128, (2 * D_INNER) / 128), 256, 0, stream>>>(
      hs16, inw16, xz, D_MODEL, D_MODEL, D_MODEL, 2 * D_INNER);

  // 3) causal depthwise conv + SiLU
  conv_silu_kernel<<<dim3(D_INNER / 256, SEQLEN, B_SZ), 256, 0, stream>>>(
      xz, convw, convb, xc, xc16);

  // 4) x_proj: xdbl[4096,192] = xc[4096,2048] @ x_proj_w^T
  wmma_gemm_xwt<64><<<dim3(NTOK / 128, XPROJ_OUT / 64), 256, 0, stream>>>(
      xc16, xpw16, xdbl, D_INNER, D_INNER, D_INNER, XPROJ_OUT);
  cvt(xdbl, xdbl16, (size_t)NTOK * XPROJ_OUT);

  // 5) dt_proj: dtraw[4096,2048] = xdbl[:, :64] @ dt_proj_w^T   (lda = 192)
  wmma_gemm_xwt<128><<<dim3(NTOK / 128, D_INNER / 128), 256, 0, stream>>>(
      xdbl16, dtw16, dtraw, DT_RANK, XPROJ_OUT, DT_RANK, D_INNER);

  // 6) dt = softplus(dtraw + bias), in place
  softplus_bias_kernel<<<4096, 256, 0, stream>>>(dtraw, dtb, NTOK * D_INNER);

  // 7) selective scan
  scan_kernel<<<dim3(D_INNER / 8, B_SZ), 256, 0, stream>>>(xdbl, dtraw, xc, Alog, yb);

  // 8) gate: (y + x*D) * silu(z) -> f16
  gate_kernel<<<4096, 256, 0, stream>>>(yb, xc, Dv, xz, g16, NTOK * D_INNER);

  // 9) out_proj: out[4096,1024] = g[4096,2048] @ out_proj_w^T
  wmma_gemm_xwt<128><<<dim3(NTOK / 128, D_MODEL / 128), 256, 0, stream>>>(
      g16, outw16, out, D_INNER, D_INNER, D_INNER, D_MODEL);
}
